// GatedGraphConv_9826885173948
// MI455X (gfx1250) — matrix operand order repaired and checked by reading, not compiled, not run on hardware
//
#include <hip/hip_runtime.h>

// ---------------------------------------------------------------------------
// GatedGraphConv on MI455X (gfx1250, wave32, WMMA bf16 16x16x32, async LDS)
// ---------------------------------------------------------------------------
#define N_NODES  50000
#define N_EDGES  800000
#define DIM      128
#define N_STEPS  3
#define N_ETYPES 4
#define STRIPS   (N_NODES / 16)         // 3125 exact
#define WSTRIDE  136                    // 128 + 8 pad -> conflict-free ds_load_b128

typedef __attribute__((ext_vector_type(16))) __bf16 v16bf;
typedef __attribute__((ext_vector_type(8)))  __bf16 bh8;    // 16 bytes
typedef __attribute__((ext_vector_type(4)))  __bf16 bh4;    // 8 bytes
typedef __attribute__((ext_vector_type(8)))  float  v8f;
typedef int v4i __attribute__((vector_size(16)));           // int4 (16 bytes)

union FragU { v16bf v; bh8 h[2]; };

// --------------------------- async global->LDS -----------------------------
#if defined(__has_builtin)
#  if __has_builtin(__builtin_amdgcn_global_load_async_to_lds_b128)
#    define HAVE_ASYNC_LDS 1
#  endif
#endif
#ifndef HAVE_ASYNC_LDS
#  define HAVE_ASYNC_LDS 0
#endif

#if HAVE_ASYNC_LDS
typedef v4i __attribute__((address_space(1)))* g_v4i_ptr;   // global
typedef v4i __attribute__((address_space(3)))* l_v4i_ptr;   // LDS
#endif

__device__ __forceinline__ void async_copy16(const void* g, void* l) {
#if HAVE_ASYNC_LDS
  __builtin_amdgcn_global_load_async_to_lds_b128(
      (g_v4i_ptr)(uintptr_t)g, (l_v4i_ptr)(uintptr_t)l, 0, 0);
#else
  *(bh8*)l = *(const bh8*)g;
#endif
}

__device__ __forceinline__ void wait_async_lds() {
#if HAVE_ASYNC_LDS
#  if __has_builtin(__builtin_amdgcn_s_wait_asynccnt)
  __builtin_amdgcn_s_wait_asynccnt(0);
#  else
  asm volatile("s_wait_asynccnt 0x0" ::: "memory");
#  endif
#endif
}

// ---------------------------------------------------------------------------
// fp32 -> bf16 elementwise convert (x4 vectorized; n must be multiple of 4)
// ---------------------------------------------------------------------------
__global__ __launch_bounds__(256) void cvt_bf16_kernel(
    const float* __restrict__ in, __bf16* __restrict__ out, int n) {
  int i = (blockIdx.x * 256 + threadIdx.x) * 4;
  if (i < n) {
    float4 v = *(const float4*)(in + i);
    bh4 o = { (__bf16)v.x, (__bf16)v.y, (__bf16)v.z, (__bf16)v.w };
    *(bh4*)(out + i) = o;
  }
}

__global__ __launch_bounds__(256) void zero_f32_kernel(float* __restrict__ p, int n) {
  int i = (blockIdx.x * 256 + threadIdx.x) * 4;
  if (i < n) *(float4*)(p + i) = make_float4(0.f, 0.f, 0.f, 0.f);
}

// ---------------------------------------------------------------------------
// T[k][n][:] = bf16( h[n] @ W_k^T + b_k )
// grid: (ceil(STRIPS/8), N_ETYPES), block: 256. W_k staged in LDS (async),
// shared by 8 waves; B fragments come from conflict-free ds_load_b128.
// ---------------------------------------------------------------------------
__global__ __launch_bounds__(256) void typed_gemm_kernel(
    const __bf16* __restrict__ hbf,      // [N_NODES][DIM]
    const __bf16* __restrict__ Wbf,      // [N_ETYPES][DIM][DIM]
    const float*  __restrict__ b_lin,    // [N_ETYPES][DIM]
    __bf16*       __restrict__ T) {      // [N_ETYPES][N_NODES][DIM]
  __shared__ __bf16 Wlds[DIM * WSTRIDE];            // 34,816 B
  const int tid = threadIdx.x;
  const int et  = blockIdx.y;

  {  // cooperative stage of W_k: 2048 x 16B chunks, 8 per thread
    const __bf16* Wg = Wbf + (size_t)et * DIM * DIM;
#pragma unroll
    for (int i = 0; i < 8; ++i) {
      const int c = tid + i * 256;
      const int row = c >> 4, col = (c & 15) * 8;
      async_copy16(Wg + row * DIM + col, &Wlds[row * WSTRIDE + col]);
    }
  }
  wait_async_lds();
  __syncthreads();

  const int wave = tid >> 5, lane = tid & 31;
  int strip = blockIdx.x * 8 + wave;
  const bool active = strip < STRIPS;
  if (!active) strip = STRIPS - 1;                  // clamp; keep EXEC full
  const int m0 = strip * 16;
  const int sub = lane >> 4, r = lane & 15;

  // A fragments: 4 chunks of K=32 resident in VGPRs, reused for 8 N-tiles
  v16bf afrag[4];
  const __bf16* arow = hbf + (size_t)(m0 + r) * DIM;
#pragma unroll
  for (int c = 0; c < 4; ++c) {
    const int k0 = c * 32;
    FragU u;
    u.h[0] = *(const bh8*)(arow + k0 + sub * 8);
    u.h[1] = *(const bh8*)(arow + k0 + 16 + sub * 8);
    afrag[c] = u.v;
  }

  const float* bias = b_lin + et * DIM;
  __bf16* Tk = T + ((size_t)et * N_NODES + m0) * DIM;

#pragma unroll
  for (int nt = 0; nt < 8; ++nt) {
    const int n0 = nt * 16;
    v8f acc = {};
#pragma unroll
    for (int c = 0; c < 4; ++c) {
      const int k0 = c * 32;
      FragU b;
      const __bf16* p = &Wlds[(n0 + r) * WSTRIDE + k0 + sub * 8];
      b.h[0] = *(const bh8*)p;
      b.h[1] = *(const bh8*)(p + 16);
      acc = __builtin_amdgcn_wmma_f32_16x16x32_bf16(
          false, afrag[c], false, b.v, (short)0, acc, false, false);
    }
    if (active) {
      const float bv = bias[n0 + r];
#pragma unroll
      for (int i = 0; i < 8; ++i) {
        const int m = sub * 8 + i;
        Tk[(size_t)m * DIM + n0 + r] = (__bf16)(acc[i] + bv);
      }
    }
  }
}

// ---------------------------------------------------------------------------
// a[dst[e]] += T[etype[e]][src[e]]  — 32 lanes per edge, 4 floats per lane,
// single 8-byte bf16x4 load, bit-shift bf16->f32, coalesced f32 atomics.
// ---------------------------------------------------------------------------
__global__ __launch_bounds__(256) void scatter_kernel(
    const __bf16* __restrict__ T,
    const int* __restrict__ src, const int* __restrict__ dst,
    const int* __restrict__ et, float* __restrict__ a) {
  const int idx = blockIdx.x * 256 + threadIdx.x;
  const int e = idx >> 5;
  if (e >= N_EDGES) return;
  const int g = (idx & 31) * 4;
  const int s = src[e], d = dst[e], k = et[e];
  const uint2 tv = *(const uint2*)(T + ((size_t)k * N_NODES + s) * DIM + g);
  float* ap = a + (size_t)d * DIM + g;
  atomicAdd(ap + 0, __uint_as_float(tv.x << 16));
  atomicAdd(ap + 1, __uint_as_float(tv.x & 0xffff0000u));
  atomicAdd(ap + 2, __uint_as_float(tv.y << 16));
  atomicAdd(ap + 3, __uint_as_float(tv.y & 0xffff0000u));
}

// ---------------------------------------------------------------------------
// Fused GRU cell. One wave per 16-node strip; a/h A-fragments in VGPRs.
// Per 16-column group: the 96 needed weight rows (r,z,n of wih & whh) are
// staged to LDS (async) once per block and shared by all 8 waves.
// ---------------------------------------------------------------------------
__device__ __forceinline__ float fast_sigmoid(float x) {
  return 1.0f / (1.0f + __expf(-x));
}
__device__ __forceinline__ float fast_tanh(float x) {
  float e = __expf(2.0f * x);
  return 1.0f - 2.0f / (e + 1.0f);
}

__global__ __launch_bounds__(256) void gru_kernel(
    const __bf16* __restrict__ abf,     // [N][DIM] aggregated messages (bf16)
    const __bf16* __restrict__ hbf,     // [N][DIM] hidden (bf16)
    const float*  __restrict__ h_in,    // [N][DIM] hidden (f32)
    const __bf16* __restrict__ wih,     // [3*DIM][DIM]
    const __bf16* __restrict__ whh,     // [3*DIM][DIM]
    const float*  __restrict__ b_ih,    // [3*DIM]
    const float*  __restrict__ b_hh,    // [3*DIM]
    float*        __restrict__ h_out) { // [N][DIM]
  __shared__ __bf16 Sl[96 * WSTRIDE];               // 26,112 B
  const int tid = threadIdx.x;
  const int wave = tid >> 5, lane = tid & 31;
  int strip = blockIdx.x * 8 + wave;
  const bool active = strip < STRIPS;
  if (!active) strip = STRIPS - 1;                  // clamp; all waves barrier
  const int m0 = strip * 16;
  const int sub = lane >> 4, r = lane & 15;

  v16bf afr[4], hfr[4];
  {
    const __bf16* ar = abf + (size_t)(m0 + r) * DIM;
    const __bf16* hp = hbf + (size_t)(m0 + r) * DIM;
#pragma unroll
    for (int c = 0; c < 4; ++c) {
      const int k0 = c * 32;
      FragU ua, uh;
      ua.h[0] = *(const bh8*)(ar + k0 + sub * 8);
      ua.h[1] = *(const bh8*)(ar + k0 + 16 + sub * 8);
      uh.h[0] = *(const bh8*)(hp + k0 + sub * 8);
      uh.h[1] = *(const bh8*)(hp + k0 + 16 + sub * 8);
      afr[c] = ua.v;
      hfr[c] = uh.v;
    }
  }

  for (int nt = 0; nt < 8; ++nt) {
    const int n0 = nt * 16;

    // stage 96 weight rows (wih r/z/n then whh r/z/n), 1536 x 16B chunks
#pragma unroll
    for (int i = 0; i < 6; ++i) {
      const int c = tid + i * 256;
      const int rowid = c >> 4, col = (c & 15) * 8;   // rowid: 0..95
      const int mat = rowid / 48, gr = rowid % 48;    // gr = gate*16 + rr
      const __bf16* sp = (mat ? whh : wih) +
          ((size_t)(gr >> 4) * DIM + n0 + (gr & 15)) * DIM + col;
      async_copy16(sp, &Sl[rowid * WSTRIDE + col]);
    }
    wait_async_lds();
    __syncthreads();

    v8f gi[3], gh[3];
#pragma unroll
    for (int g = 0; g < 3; ++g) {
      v8f ai = {}, hi = {};
#pragma unroll
      for (int c = 0; c < 4; ++c) {
        const int k0 = c * 32;
        FragU bi, bh;
        const __bf16* pi = &Sl[(g * 16 + r) * WSTRIDE + k0 + sub * 8];
        const __bf16* ph = &Sl[(48 + g * 16 + r) * WSTRIDE + k0 + sub * 8];
        bi.h[0] = *(const bh8*)pi;  bi.h[1] = *(const bh8*)(pi + 16);
        bh.h[0] = *(const bh8*)ph;  bh.h[1] = *(const bh8*)(ph + 16);
        ai = __builtin_amdgcn_wmma_f32_16x16x32_bf16(
            false, afr[c], false, bi.v, (short)0, ai, false, false);
        hi = __builtin_amdgcn_wmma_f32_16x16x32_bf16(
            false, hfr[c], false, bh.v, (short)0, hi, false, false);
      }
      gi[g] = ai;
      gh[g] = hi;
    }
    __syncthreads();   // all reads of Sl done before next iteration's stage

    if (active) {
      const int col = n0 + r;
      const float bi_r = b_ih[col],           bh_r = b_hh[col];
      const float bi_z = b_ih[DIM + col],     bh_z = b_hh[DIM + col];
      const float bi_n = b_ih[2 * DIM + col], bh_n = b_hh[2 * DIM + col];
#pragma unroll
      for (int i = 0; i < 8; ++i) {
        const float ir  = gi[0][i] + bi_r, hr_ = gh[0][i] + bh_r;
        const float iz  = gi[1][i] + bi_z, hz  = gh[1][i] + bh_z;
        const float inn = gi[2][i] + bi_n, hn  = gh[2][i] + bh_n;
        const float rg = fast_sigmoid(ir + hr_);
        const float zg = fast_sigmoid(iz + hz);
        const float ng = fast_tanh(inn + rg * hn);
        const size_t off = (size_t)(m0 + sub * 8 + i) * DIM + col;
        h_out[off] = (1.0f - zg) * ng + zg * h_in[off];
      }
    }
  }
}

// ---------------------------------------------------------------------------
// Host-side orchestration
// ---------------------------------------------------------------------------
static inline size_t align256(size_t x) { return (x + 255) & ~(size_t)255; }

extern "C" void kernel_launch(void* const* d_in, const int* in_sizes, int n_in,
                              void* d_out, int out_size, void* d_ws, size_t ws_size,
                              hipStream_t stream) {
  const float* feat  = (const float*)d_in[0];
  const float* W_lin = (const float*)d_in[1];
  const float* b_lin = (const float*)d_in[2];
  const float* w_ih  = (const float*)d_in[3];
  const float* w_hh  = (const float*)d_in[4];
  const float* b_ih  = (const float*)d_in[5];
  const float* b_hh  = (const float*)d_in[6];
  const int*   src   = (const int*)d_in[7];
  const int*   dst   = (const int*)d_in[8];
  const int*   etyp  = (const int*)d_in[9];
  float* out = (float*)d_out;

  char* w = (char*)d_ws;
  __bf16* Wbf   = (__bf16*)w;  w += align256((size_t)N_ETYPES * DIM * DIM * 2);
  __bf16* wihbf = (__bf16*)w;  w += align256((size_t)3 * DIM * DIM * 2);
  __bf16* whhbf = (__bf16*)w;  w += align256((size_t)3 * DIM * DIM * 2);
  __bf16* hbf   = (__bf16*)w;  w += align256((size_t)N_NODES * DIM * 2);
  __bf16* abf   = (__bf16*)w;  w += align256((size_t)N_NODES * DIM * 2);
  float*  aggr  = (float*)w;   w += align256((size_t)N_NODES * DIM * 4);
  __bf16* T     = (__bf16*)w;  w += align256((size_t)N_ETYPES * N_NODES * DIM * 2);
  float*  h1    = (float*)w;   w += align256((size_t)N_NODES * DIM * 4);
  float*  h2    = (float*)w;   w += align256((size_t)N_NODES * DIM * 4);

  const int nfeat = N_NODES * DIM;
  const dim3 blk(256);

  cvt_bf16_kernel<<<dim3((N_ETYPES * DIM * DIM / 4 + 255) / 256), blk, 0, stream>>>(
      W_lin, Wbf, N_ETYPES * DIM * DIM);
  cvt_bf16_kernel<<<dim3((3 * DIM * DIM / 4 + 255) / 256), blk, 0, stream>>>(
      w_ih, wihbf, 3 * DIM * DIM);
  cvt_bf16_kernel<<<dim3((3 * DIM * DIM / 4 + 255) / 256), blk, 0, stream>>>(
      w_hh, whhbf, 3 * DIM * DIM);

  const dim3 gemm_grid((STRIPS + 7) / 8, N_ETYPES);
  const dim3 gru_grid((STRIPS + 7) / 8);
  const dim3 scat_grid(((size_t)N_EDGES * 32 + 255) / 256);
  const dim3 cvt_grid((nfeat / 4 + 255) / 256);

  for (int step = 0; step < N_STEPS; ++step) {
    const float* h_cur = (step == 0) ? feat : ((step == 1) ? h1 : h2);
    float* h_nxt = (step == 0) ? h1 : ((step == 1) ? h2 : out);

    cvt_bf16_kernel<<<cvt_grid, blk, 0, stream>>>(h_cur, hbf, nfeat);
    typed_gemm_kernel<<<gemm_grid, blk, 0, stream>>>(hbf, Wbf, b_lin, T);
    zero_f32_kernel<<<cvt_grid, blk, 0, stream>>>(aggr, nfeat);
    scatter_kernel<<<scat_grid, blk, 0, stream>>>(T, src, dst, etyp, aggr);
    cvt_bf16_kernel<<<cvt_grid, blk, 0, stream>>>(aggr, abf, nfeat);
    gru_kernel<<<gru_grid, blk, 0, stream>>>(abf, hbf, h_cur, wihbf, whhbf,
                                             b_ih, b_hh, h_nxt);
  }
}
